// AtomUpdateBlock_35485019799896
// MI455X (gfx1250) — matrix operand order repaired and checked by reading, not compiled, not run on hardware
//
#include <hip/hip_runtime.h>

// ---------------------------------------------------------------------------
// GemNet AtomUpdateBlock for MI455X (gfx1250, wave32, WMMA).
//  Kernel 1: zero the [nAtoms,128] scatter accumulator in d_ws.
//  Kernel 2: edge kernel  x = m * (rbf @ W_rbf.T), atomic scatter-add -> x2.
//            (memory-bound: 580MB HBM stream is the whole-problem floor)
//  Kernel 3: fused atom MLP, 5x (128x128 GEMM) via V_WMMA_F32_16X16X32_F16
//            (f16 in / fp32 accumulate -> 8x the dense rate of f32 WMMA),
//            scaled-SiLU + residual combines in fp32, all LDS-resident.
//            Layer schedule is fully static so x-tile accesses stay in
//            addrspace(3) and lower to ds_load_b128 (no flat fallback).
// ---------------------------------------------------------------------------

typedef __attribute__((ext_vector_type(8)))  float     f32x8;
typedef __attribute__((ext_vector_type(4)))  _Float16  h4;
typedef __attribute__((ext_vector_type(8)))  _Float16  h8;
typedef __attribute__((ext_vector_type(16))) _Float16  h16;

#define DE 128
#define DRBF 16
#define PITCH 136                 // f16 LDS row pitch: 272B, 16B-aligned, bank-spread
#define TILE_H (128 * PITCH)      // f16 elements per 128-row tile
#define INV_SCALE_SILU 1.6666666666666667f
#define INV_SQRT2 0.7071067811865476f

__device__ __forceinline__ float scaled_silu_f(float v) {
  return v * (1.0f / (1.0f + __expf(-v))) * INV_SCALE_SILU;
}

// ---------------------------------------------------------------------------
__global__ void __launch_bounds__(256) zero_kernel(float* __restrict__ p, size_t n) {
  size_t i = (size_t)blockIdx.x * blockDim.x + threadIdx.x;
  const size_t stride = (size_t)gridDim.x * blockDim.x;
  for (; i < n; i += stride) p[i] = 0.0f;
}

// ---------------------------------------------------------------------------
// Edge phase: each thread owns one channel of one edge slot; W_rbf row in LDS.
// Scatter via global_atomic_add_f32 (no-return -> STOREcnt); the 51MB target
// stays resident in the 192MB L2.
__global__ void __launch_bounds__(256) edge_scatter_kernel(
    const float* __restrict__ m, const float* __restrict__ rbf,
    const int* __restrict__ id_j, const float* __restrict__ W_rbf,
    float* __restrict__ x2, int E) {
  __shared__ float wr[DE * (DRBF + 1)];
  const int tid = threadIdx.x;
  for (int i = tid; i < DE * DRBF; i += 256) {
    int r = i >> 4, c = i & 15;
    wr[r * (DRBF + 1) + c] = W_rbf[i];
  }
  __syncthreads();

  const int c = tid & 127;   // output channel
  const int s = tid >> 7;    // edge slot within block (0/1)
  const float* wrow = &wr[c * (DRBF + 1)];
  const int estride = gridDim.x * 2;

  for (int e = blockIdx.x * 2 + s; e < E; e += estride) {
    if (e + estride < E)     // keep the HBM read stream ahead (global_prefetch_b8)
      __builtin_prefetch(&m[(size_t)(e + estride) * DE + c], 0, 1);
    const float* r = &rbf[(size_t)e * DRBF];   // broadcast load (one line per wave)
    float dot = 0.0f;
#pragma unroll
    for (int k = 0; k < DRBF; ++k) dot = __fmaf_rn(r[k], wrow[k], dot);
    const float val = m[(size_t)e * DE + c] * dot;
    atomicAdd(&x2[(size_t)id_j[e] * DE + c], val);
  }
}

// ---------------------------------------------------------------------------
// WMMA f16 16x16x32 fragment gathers from padded f16 LDS tiles.
// A (16x32): lane row = m0+l15; K chunks {k0+8hi..+7} and {k0+16+8hi..+7}.
// B (32x16) for x@W.T (W stored [out,in]): lane row = n0+l15 of W;
//           K chunks {k0+8hi..+7} and {k0+16+8hi..+7} (same as A).
__device__ __forceinline__ h16 ldA(const _Float16* buf, int row, int k0, int hi8) {
  const _Float16* p = &buf[row * PITCH + k0 + hi8];
  h8 lo = *(const h8*)p;
  h8 hi = *(const h8*)(p + 16);
  return __builtin_shufflevector(lo, hi, 0, 1, 2, 3, 4, 5, 6, 7,
                                 8, 9, 10, 11, 12, 13, 14, 15);
}

__device__ __forceinline__ h16 ldB(const _Float16* buf, int row, int k0, int hi8) {
  const _Float16* p = &buf[row * PITCH + k0 + hi8];
  h8 lo = *(const h8*)p;
  h8 hi = *(const h8*)(p + 16);
  return __builtin_shufflevector(lo, hi, 0, 1, 2, 3, 4, 5, 6, 7,
                                 8, 9, 10, 11, 12, 13, 14, 15);
}

// Stage one 128x128 fp32 weight matrix into a padded f16 LDS tile (coalesced).
__device__ __forceinline__ void load_weight_lds(_Float16* wbuf,
                                                const float* __restrict__ W,
                                                int tid) {
  const float4* g = (const float4*)W;
#pragma unroll
  for (int i = 0; i < 16; ++i) {
    int idx4 = i * 256 + tid;                    // 0..4095
    int row = idx4 >> 5;
    int c4 = (idx4 & 31) << 2;
    const float4 v = g[idx4];
    h4 o = {(_Float16)v.x, (_Float16)v.y, (_Float16)v.z, (_Float16)v.w};
    *(h4*)&wbuf[row * PITCH + c4] = o;
  }
}

// Epilogue modes (compile-time so LDS pointers stay static).
#define EPI_SILU 0      // dst_lds = silu
#define EPI_RES  1      // dst_lds = (res + silu) * invsqrt2
#define EPI_OUT  2      // global fp32 = (res + silu) * invsqrt2 (final layer)

// One dense layer over the 128-atom tile: this wave computes a 16x128 strip.
template <int MODE>
__device__ __forceinline__ void dense_layer(const _Float16* src,
                                            const _Float16* wbuf,
                                            _Float16* dst,
                                            const _Float16* res,
                                            float* __restrict__ gout,
                                            int base, int nAtoms,
                                            int m0, int l15, int hi8) {
  f32x8 acc[8];
#pragma unroll
  for (int t = 0; t < 8; ++t)
    acc[t] = (f32x8){0.f, 0.f, 0.f, 0.f, 0.f, 0.f, 0.f, 0.f};

#pragma unroll
  for (int k = 0; k < 4; ++k) {                  // K = 128 in steps of 32
    const int k0 = 32 * k;
    const h16 a = ldA(src, m0 + l15, k0, hi8);
#pragma unroll
    for (int t = 0; t < 8; ++t) {                // N = 128 in 16-wide tiles
      const h16 b = ldB(wbuf, 16 * t + l15, k0, hi8);
      acc[t] = __builtin_amdgcn_wmma_f32_16x16x32_f16(
          /*neg_a=*/false, a, /*neg_b=*/false, b,
          /*c_mod=*/(short)0, acc[t], /*reuse_a=*/false, /*reuse_b=*/false);
    }
  }

  // D layout (16x16 f32, 8 VGPRs): VGPR v -> row m0+v+8hi, col n0+l15.
#pragma unroll
  for (int t = 0; t < 8; ++t) {
    const int n0 = 16 * t;
#pragma unroll
    for (int v = 0; v < 8; ++v) {
      const int row = m0 + v + hi8;
      const int col = n0 + l15;
      float val = scaled_silu_f(acc[t][v]);
      if (MODE != EPI_SILU)
        val = ((float)res[row * PITCH + col] + val) * INV_SQRT2;
      if (MODE == EPI_OUT) {                     // final layer: fp32 to global
        const int grow = base + row;
        if (grow < nAtoms) gout[(size_t)grow * DE + col] = val;
      } else {
        dst[row * PITCH + col] = (_Float16)val;
      }
    }
  }
}

// ---------------------------------------------------------------------------
// Fused atom MLP: x = silu(x2*scale @ W1.T); 2x residual blocks.
// LDS: f16 weight buffer + 3 ping-pong f16 x tiles = 4 * 128*136*2 B = 136 KB
// -> 2 workgroups per 320KB WGP (4 waves/SIMD).
__global__ void __launch_bounds__(256) atom_mlp_kernel(
    const float* __restrict__ x2, const float* __restrict__ scale,
    const float* __restrict__ W1, const float* __restrict__ W_res,
    float* __restrict__ out, int nAtoms) {
  __shared__ _Float16 wbuf[TILE_H];
  __shared__ _Float16 xa[TILE_H];
  __shared__ _Float16 xb[TILE_H];
  __shared__ _Float16 xc[TILE_H];

  const int tid = threadIdx.x;
  const int lane = tid & 31;
  const int l15 = lane & 15;
  const int hi8 = (lane >> 4) << 3;              // 8 * (lane >= 16)
  const int m0 = (tid >> 5) * 16;                // this wave's 16-row block
  const int base = blockIdx.x * 128;
  const float sc = scale[0];

  // Stage x tile (ScaleFactor applied, fp32 -> f16); clamp ragged rows.
#pragma unroll
  for (int i = 0; i < 16; ++i) {
    int idx4 = i * 256 + tid;
    int r = idx4 >> 5;
    int c4 = (idx4 & 31) << 2;
    int grow = base + r;
    if (grow >= nAtoms) grow = nAtoms - 1;
    const float4 v = *(const float4*)&x2[(size_t)grow * DE + c4];
    h4 o = {(_Float16)(v.x * sc), (_Float16)(v.y * sc),
            (_Float16)(v.z * sc), (_Float16)(v.w * sc)};
    *(h4*)&xa[r * PITCH + c4] = o;
  }

  const float* Wr0 = W_res;
  const float* Wr1 = W_res + 16384;
  const float* Wr2 = W_res + 2 * 16384;
  const float* Wr3 = W_res + 3 * 16384;

  // Static schedule: L0: xa->xb; R0: xb->xc, xc->xa(+xb); R1: xa->xc, xc->out(+xa).
  __syncthreads();
  load_weight_lds(wbuf, W1, tid);
  for (int i = tid; i < 512; i += 256) __builtin_prefetch(Wr0 + (size_t)i * 32, 0, 1);
  __syncthreads();
  dense_layer<EPI_SILU>(xa, wbuf, xb, nullptr, nullptr, base, nAtoms, m0, l15, hi8);

  __syncthreads();
  load_weight_lds(wbuf, Wr0, tid);
  for (int i = tid; i < 512; i += 256) __builtin_prefetch(Wr1 + (size_t)i * 32, 0, 1);
  __syncthreads();
  dense_layer<EPI_SILU>(xb, wbuf, xc, nullptr, nullptr, base, nAtoms, m0, l15, hi8);

  __syncthreads();
  load_weight_lds(wbuf, Wr1, tid);
  for (int i = tid; i < 512; i += 256) __builtin_prefetch(Wr2 + (size_t)i * 32, 0, 1);
  __syncthreads();
  dense_layer<EPI_RES>(xc, wbuf, xa, xb, nullptr, base, nAtoms, m0, l15, hi8);

  __syncthreads();
  load_weight_lds(wbuf, Wr2, tid);
  for (int i = tid; i < 512; i += 256) __builtin_prefetch(Wr3 + (size_t)i * 32, 0, 1);
  __syncthreads();
  dense_layer<EPI_SILU>(xa, wbuf, xc, nullptr, nullptr, base, nAtoms, m0, l15, hi8);

  __syncthreads();
  load_weight_lds(wbuf, Wr3, tid);
  __syncthreads();
  dense_layer<EPI_OUT>(xc, wbuf, nullptr, xa, out, base, nAtoms, m0, l15, hi8);
}

// ---------------------------------------------------------------------------
extern "C" void kernel_launch(void* const* d_in, const int* in_sizes, int n_in,
                              void* d_out, int out_size, void* d_ws, size_t ws_size,
                              hipStream_t stream) {
  // setup_inputs order: nAtoms, m, rbf, id_j, W_rbf, scale, W1, W_res
  const float* m     = (const float*)d_in[1];
  const float* rbf   = (const float*)d_in[2];
  const int*   id_j  = (const int*)d_in[3];
  const float* W_rbf = (const float*)d_in[4];
  const float* scale = (const float*)d_in[5];
  const float* W1    = (const float*)d_in[6];
  const float* W_res = (const float*)d_in[7];
  float* out = (float*)d_out;
  float* x2  = (float*)d_ws;                     // [nAtoms,128] scatter accumulator

  const int E = in_sizes[1] / DE;                // m is [E,128]
  const int nAtoms = out_size / DE;              // out is [nAtoms,128]
  const size_t nx2 = (size_t)nAtoms * DE;

  zero_kernel<<<1024, 256, 0, stream>>>(x2, nx2);
  edge_scatter_kernel<<<8192, 256, 0, stream>>>(m, rbf, id_j, W_rbf, x2, E);

  const int nblk = (nAtoms + 127) / 128;
  atom_mlp_kernel<<<nblk, 256, 0, stream>>>(x2, scale, W1, W_res, out, nAtoms);
}
